// SelfAttention_10136122819252
// MI455X (gfx1250) — matrix operand order repaired and checked by reading, not compiled, not run on hardware
//
#include <hip/hip_runtime.h>
#include <hip/hip_bf16.h>

typedef unsigned short u16;
typedef __attribute__((ext_vector_type(16))) __bf16 v16bf;
typedef __attribute__((ext_vector_type(8)))  float  v8f;

#define BB 16
#define SS 1024
#define EE 512
#define DD 512
#define NEGINF (-1e10f)

__device__ __forceinline__ u16 f2bf(float f) {
  unsigned int u = __float_as_uint(f);
  u += 0x7FFFu + ((u >> 16) & 1u);
  return (u16)(u >> 16);
}

// ---------------------------------------------------------------------------
// 1) Embedding gather: x_bf16[s, e] = bf16(embed_table[inputs[s], e])
// ---------------------------------------------------------------------------
__global__ __launch_bounds__(256) void embed_kernel(
    const int* __restrict__ tok, const float* __restrict__ tab,
    u16* __restrict__ xbf) {
  size_t i = (size_t)blockIdx.x * 256 + threadIdx.x;  // over B*S*E
  int s = (int)(i >> 9);                              // /512
  int e = (int)(i & 511);
  int t = tok[s];
  xbf[i] = f2bf(tab[(size_t)t * EE + e]);
}

// ---------------------------------------------------------------------------
// 2) Weight transpose+convert: WT[n][k] = bf16(W[k][n]); 512x512
// ---------------------------------------------------------------------------
__global__ __launch_bounds__(256) void wtrans_kernel(
    const float* __restrict__ W, u16* __restrict__ WT) {
  int i = blockIdx.x * 256 + threadIdx.x;  // over 512*512
  int n = i >> 9;
  int k = i & 511;
  WT[(size_t)n * EE + k] = f2bf(W[(size_t)k * DD + n]);
}

// ---------------------------------------------------------------------------
// 3) Per-batch bf16 transpose: Vt[b][d][s] = V[b][s][d]
// ---------------------------------------------------------------------------
__global__ __launch_bounds__(256) void transpose_v_kernel(
    const u16* __restrict__ V, u16* __restrict__ Vt) {
  __shared__ u16 tile[32][33];
  int b = blockIdx.z;
  const u16* Vb = V + (size_t)b * SS * DD;
  u16* Vtb = Vt + (size_t)b * DD * SS;
  int d0 = blockIdx.x * 32, s0 = blockIdx.y * 32;
  int tx = threadIdx.x, ty = threadIdx.y;  // (32,8)
#pragma unroll
  for (int i = 0; i < 4; i++)
    tile[ty + i * 8][tx] = Vb[(size_t)(s0 + ty + i * 8) * DD + d0 + tx];
  __syncthreads();
#pragma unroll
  for (int i = 0; i < 4; i++)
    Vtb[(size_t)(d0 + ty + i * 8) * SS + s0 + tx] = tile[tx][ty + i * 8];
}

// ---------------------------------------------------------------------------
// 4) Generic bf16 WMMA GEMM: C[m,n] = sum_k A[m,k] * Bt[n,k] (+bias)
//    4 waves, 64x64 macro-tile, each wave 32x32 (2x2 fragments), Ktile=32.
//    Double-buffered LDS staging; one barrier per K-step.
//    EPI 0: C = acc (+bias)
//    EPI 1: C = tanh(acc + bias)
//    EPI 2: score epilogue: dep gather, scale, mask
//    HASF / HASB / HASBIAS: compile-time output & bias selection.
// ---------------------------------------------------------------------------
#define LDSP 40  // padded row stride (ushorts), 80B keeps 16B alignment

template <int EPI, bool HASF, bool HASB, bool HASBIAS>
__global__ __launch_bounds__(128) void gemm_bf16_kernel(
    const u16* __restrict__ A, long long strideA,
    const u16* __restrict__ Bt, long long strideB,
    const float* __restrict__ bias,
    float* __restrict__ outF, long long strideF,
    u16* __restrict__ outB, long long strideOB,
    int M, int N, int K, int lda, int ldb, int ldc,
    const int* __restrict__ dep, const float* __restrict__ depTab) {
  __shared__ u16 As[2][64 * LDSP];
  __shared__ u16 Bs[2][64 * LDSP];
  const int tid = threadIdx.x;
  const int wid = tid >> 5;
  const int lane = tid & 31;
  const int hl = lane >> 4;    // half-wave select
  const int l15 = lane & 15;
  const int wm = wid >> 1;     // 2x2 wave grid
  const int wn = wid & 1;
  const int row0 = blockIdx.y * 64;
  const int col0 = blockIdx.x * 64;
  const int bz = blockIdx.z;
  A += (long long)bz * strideA;
  Bt += (long long)bz * strideB;
  if (HASF) outF += (long long)bz * strideF;
  if (HASB) outB += (long long)bz * strideOB;
  if (EPI == 2) dep += (size_t)bz * (size_t)M * (size_t)N;

  // Per-thread staging addresses (two 16B chunks per tile per thread).
  const int c0 = tid, c1 = tid + 128;
  const int r0 = c0 >> 2, cc0 = c0 & 3;
  const int r1 = c1 >> 2, cc1 = c1 & 3;
  const u16* aSrc0 = A + (size_t)(row0 + r0) * lda + cc0 * 8;
  const u16* aSrc1 = A + (size_t)(row0 + r1) * lda + cc1 * 8;
  const u16* bSrc0 = Bt + (size_t)(col0 + r0) * ldb + cc0 * 8;
  const u16* bSrc1 = Bt + (size_t)(col0 + r1) * ldb + cc1 * 8;
  const int aDst0 = r0 * LDSP + cc0 * 8, aDst1 = r1 * LDSP + cc1 * 8;

  auto stage = [&](int kt, int buf) {
    *(uint4*)(&As[buf][aDst0]) = *(const uint4*)(aSrc0 + kt);
    *(uint4*)(&As[buf][aDst1]) = *(const uint4*)(aSrc1 + kt);
    *(uint4*)(&Bs[buf][aDst0]) = *(const uint4*)(bSrc0 + kt);
    *(uint4*)(&Bs[buf][aDst1]) = *(const uint4*)(bSrc1 + kt);
  };

  v8f acc[2][2];
#pragma unroll
  for (int i = 0; i < 2; i++)
#pragma unroll
    for (int j = 0; j < 2; j++)
#pragma unroll
      for (int r = 0; r < 8; r++) acc[i][j][r] = 0.f;

  const int nst = K >> 5;  // K/32
  stage(0, 0);
  for (int t = 0; t < nst; ++t) {
    __syncthreads();
    if (t + 1 < nst) stage((t + 1) << 5, (t + 1) & 1);
    const int buf = t & 1;

    // A fragment (16x32 bf16): lanes 0-15 row M=l15 hold K 0-7 & 16-23,
    // lanes 16-31 hold K 8-15 & 24-31  -> two b128 LDS reads per lane.
    v16bf af[2];
#pragma unroll
    for (int i = 0; i < 2; i++) {
      int ra = wm * 32 + i * 16 + l15;
      int kb = hl * 8;
      union { uint4 u[2]; v16bf v; } ua;
      ua.u[0] = *(const uint4*)(&As[buf][ra * LDSP + kb]);
      ua.u[1] = *(const uint4*)(&As[buf][ra * LDSP + kb + 16]);
      af[i] = ua.v;
    }
    // B fragment (32x16 bf16): lane holds column N=l15, K 0-7 & 16-23
    // (lanes 0-15) or K 8-15 & 24-31 (lanes 16-31).
    v16bf bfr[2];
#pragma unroll
    for (int j = 0; j < 2; j++) {
      int cb = wn * 32 + j * 16 + l15;
      int kb = hl * 8;
      union { uint4 u[2]; v16bf v; } ub;
      ub.u[0] = *(const uint4*)(&Bs[buf][cb * LDSP + kb]);
      ub.u[1] = *(const uint4*)(&Bs[buf][cb * LDSP + kb + 16]);
      bfr[j] = ub.v;
    }
#pragma unroll
    for (int i = 0; i < 2; i++)
#pragma unroll
      for (int j = 0; j < 2; j++)
        acc[i][j] = __builtin_amdgcn_wmma_f32_16x16x32_bf16(
            false, af[i], false, bfr[j], (short)0, acc[i][j], false, false);
  }

  // Epilogue. C fragment layout: VGPR r -> M = r + 8*hl, N = l15.
  const int mbase = row0 + wm * 32 + hl * 8;
#pragma unroll
  for (int i = 0; i < 2; i++) {
#pragma unroll
    for (int j = 0; j < 2; j++) {
      const int m0 = mbase + i * 16;
      const int n = col0 + wn * 32 + j * 16 + l15;
      const float bia = HASBIAS ? bias[n] : 0.f;
      const size_t base = (size_t)m0 * ldc + n;
#pragma unroll
      for (int r = 0; r < 8; r++) {
        float v = acc[i][j][r] + bia;
        if (EPI == 1) v = tanhf(v);
        if (EPI == 2) {
          int dv = dep[base + (size_t)r * ldc];
          v = (dv == 0) ? NEGINF : v * depTab[dv];
        }
        if (HASF) outF[base + (size_t)r * ldc] = v;
        if (HASB) outB[base + (size_t)r * ldc] = f2bf(v);
      }
    }
  }
}

// ---------------------------------------------------------------------------
// 5) Row softmax in place over d_out weights region; emits bf16 copy.
// ---------------------------------------------------------------------------
__global__ __launch_bounds__(256) void softmax_kernel(
    float* __restrict__ w, u16* __restrict__ wbf) {
  __shared__ float red[256];
  const int tid = threadIdx.x;
  size_t base = (size_t)blockIdx.x * SS;
  float* p = w + base;
  float mx = -3.4e38f;
  for (int i = tid; i < SS; i += 256) mx = fmaxf(mx, p[i]);
  red[tid] = mx;
  __syncthreads();
  for (int s = 128; s > 0; s >>= 1) {
    if (tid < s) red[tid] = fmaxf(red[tid], red[tid + s]);
    __syncthreads();
  }
  mx = red[0];
  __syncthreads();
  float sum = 0.f;
  for (int i = tid; i < SS; i += 256) {
    float e = __expf(p[i] - mx);
    p[i] = e;
    sum += e;
  }
  red[tid] = sum;
  __syncthreads();
  for (int s = 128; s > 0; s >>= 1) {
    if (tid < s) red[tid] += red[tid + s];
    __syncthreads();
  }
  float inv = 1.f / red[0];
  for (int i = tid; i < SS; i += 256) {
    float v = p[i] * inv;
    p[i] = v;
    wbf[base + i] = f2bf(v);
  }
}

// ---------------------------------------------------------------------------
// Host launcher
// ---------------------------------------------------------------------------
extern "C" void kernel_launch(void* const* d_in, const int* in_sizes, int n_in,
                              void* d_out, int out_size, void* d_ws,
                              size_t ws_size, hipStream_t stream) {
  (void)in_sizes; (void)n_in; (void)out_size; (void)ws_size;
  const int* inputs = (const int*)d_in[0];
  const int* dependency = (const int*)d_in[1];
  const float* embed_table = (const float*)d_in[2];
  const float* dep_table = (const float*)d_in[3];
  const float* Wq = (const float*)d_in[4];
  const float* bq = (const float*)d_in[5];
  const float* Wk = (const float*)d_in[6];
  const float* bk = (const float*)d_in[7];
  const float* Wv = (const float*)d_in[8];
  const float* bv = (const float*)d_in[9];
  const float* W1 = (const float*)d_in[10];
  const float* b1 = (const float*)d_in[11];
  const float* W2 = (const float*)d_in[12];
  const float* b2 = (const float*)d_in[13];

  float* out_wm = (float*)d_out;                 // [B,S,D]
  float* out_w = out_wm + (size_t)BB * SS * DD;  // [B,S,S]

  char* ws = (char*)d_ws;
  size_t off = 0;
  auto alloc = [&](size_t bytes) {
    void* p = ws + off;
    off = (off + bytes + 255) & ~(size_t)255;
    return p;
  };
  u16* xbf = (u16*)alloc((size_t)BB * SS * EE * 2);
  u16* WqT = (u16*)alloc((size_t)EE * DD * 2);
  u16* WkT = (u16*)alloc((size_t)EE * DD * 2);
  u16* WvT = (u16*)alloc((size_t)EE * DD * 2);
  u16* W1T = (u16*)alloc((size_t)DD * DD * 2);
  u16* W2T = (u16*)alloc((size_t)DD * DD * 2);
  u16* Q = (u16*)alloc((size_t)BB * SS * DD * 2);
  u16* Kb = (u16*)alloc((size_t)BB * SS * DD * 2);
  u16* V = (u16*)alloc((size_t)BB * SS * DD * 2);
  u16* Vt = (u16*)alloc((size_t)BB * DD * SS * 2);
  u16* wbf = (u16*)alloc((size_t)BB * SS * SS * 2);
  u16* wmb = (u16*)alloc((size_t)BB * SS * DD * 2);
  u16* hb = (u16*)alloc((size_t)BB * SS * DD * 2);

  // 1) embed gather + bf16 convert
  embed_kernel<<<(BB * SS * EE) / 256, 256, 0, stream>>>(inputs, embed_table, xbf);

  // 2) weights -> bf16 transposed (N-major) layout
  wtrans_kernel<<<(EE * DD) / 256, 256, 0, stream>>>(Wq, WqT);
  wtrans_kernel<<<(EE * DD) / 256, 256, 0, stream>>>(Wk, WkT);
  wtrans_kernel<<<(EE * DD) / 256, 256, 0, stream>>>(Wv, WvT);
  wtrans_kernel<<<(DD * DD) / 256, 256, 0, stream>>>(W1, W1T);
  wtrans_kernel<<<(DD * DD) / 256, 256, 0, stream>>>(W2, W2T);

  // 3) Q/K/V projections: [16384,512] x [512,512] -> bf16
  gemm_bf16_kernel<0, false, true, true>
      <<<dim3(DD / 64, (BB * SS) / 64, 1), 128, 0, stream>>>(
          xbf, 0, WqT, 0, bq, nullptr, 0, Q, 0,
          BB * SS, DD, EE, EE, EE, DD, nullptr, nullptr);
  gemm_bf16_kernel<0, false, true, true>
      <<<dim3(DD / 64, (BB * SS) / 64, 1), 128, 0, stream>>>(
          xbf, 0, WkT, 0, bk, nullptr, 0, Kb, 0,
          BB * SS, DD, EE, EE, EE, DD, nullptr, nullptr);
  gemm_bf16_kernel<0, false, true, true>
      <<<dim3(DD / 64, (BB * SS) / 64, 1), 128, 0, stream>>>(
          xbf, 0, WvT, 0, bv, nullptr, 0, V, 0,
          BB * SS, DD, EE, EE, EE, DD, nullptr, nullptr);

  // 4) V -> Vt per batch
  transpose_v_kernel<<<dim3(DD / 32, SS / 32, BB), dim3(32, 8), 0, stream>>>(V, Vt);

  // 5) scores = (Q K^T) * dep_emb, masked -> f32 straight into d_out weights
  gemm_bf16_kernel<2, true, false, false>
      <<<dim3(SS / 64, SS / 64, BB), 128, 0, stream>>>(
          Q, (long long)SS * DD, Kb, (long long)SS * DD, nullptr,
          out_w, (long long)SS * SS, nullptr, 0,
          SS, SS, DD, DD, DD, SS, dependency, dep_table);

  // 6) softmax in place (f32) + bf16 copy
  softmax_kernel<<<BB * SS, 256, 0, stream>>>(out_w, wbf);

  // 7) wm = weights @ V  (per batch)
  gemm_bf16_kernel<0, false, true, false>
      <<<dim3(DD / 64, SS / 64, BB), 128, 0, stream>>>(
          wbf, (long long)SS * SS, Vt, (long long)DD * SS, nullptr,
          nullptr, 0, wmb, (long long)SS * DD,
          SS, DD, SS, SS, SS, DD, nullptr, nullptr);

  // 8) h = tanh(wm @ W1 + b1)
  gemm_bf16_kernel<1, false, true, true>
      <<<dim3(DD / 64, (BB * SS) / 64, 1), 128, 0, stream>>>(
          wmb, 0, W1T, 0, b1, nullptr, 0, hb, 0,
          BB * SS, DD, DD, DD, DD, DD, nullptr, nullptr);

  // 9) out = h @ W2 + b2 -> f32 d_out
  gemm_bf16_kernel<0, true, false, true>
      <<<dim3(DD / 64, (BB * SS) / 64, 1), 128, 0, stream>>>(
          hb, 0, W2T, 0, b2, out_wm, 0, nullptr, 0,
          BB * SS, DD, DD, DD, DD, DD, nullptr, nullptr);
}
